// GAU_16277926051946
// MI455X (gfx1250) — matrix operand order repaired and checked by reading, not compiled, not run on hardware
//
#include <hip/hip_runtime.h>

// ---------------------------------------------------------------------------
// GAU block for MI455X (gfx1250): bf16 WMMA pipeline, flash-style attention.
// B=8, S=2048, D=512, QK=128, HID=1024 (v/gate each), H2=2048.
// ---------------------------------------------------------------------------

typedef __attribute__((ext_vector_type(16))) __bf16 v16bf;
typedef __attribute__((ext_vector_type(8)))  float  v8f;
typedef __attribute__((ext_vector_type(8)))  __bf16 bf16x8;

union FragB { v16bf v; bf16x8 h[2]; };

__device__ __forceinline__ v8f wmma_bf16(v16bf a, v16bf b, v8f c) {
  // D = A(16x32) * B(32x16) + C(16x16 f32)
  return __builtin_amdgcn_wmma_f32_16x16x32_bf16(
      /*neg_a=*/false, a, /*neg_b=*/false, b,
      /*c_mod=*/(short)0, c, /*reuse_a=*/false, /*reuse_b=*/false);
}

// A-fragment (16x32 bf16, row m = lane&15): K = (j>>3)*16 + (lane>>4)*8 + (j&7)
__device__ __forceinline__ void load_a(FragB& f, const __bf16* row, int kt, int hl) {
  f.h[0] = *(const bf16x8*)(row + kt * 32 + hl * 8);
  f.h[1] = *(const bf16x8*)(row + kt * 32 + 16 + hl * 8);
}
// B-fragment (32x16 bf16, col n = lane&15): K = (j>>3)*16 + (lane>>4)*8 + (j&7)
__device__ __forceinline__ void load_b(FragB& f, const __bf16* row, int kt, int hl) {
  f.h[0] = *(const bf16x8*)(row + kt * 32 + hl * 8);
  f.h[1] = *(const bf16x8*)(row + kt * 32 + 16 + hl * 8);
}

// SiLU with HW reciprocal (v_exp + v_rcp), avoids IEEE-div sequence.
__device__ __forceinline__ float fast_silu(float v) {
  float e = __expf(-v);
  return v * __builtin_amdgcn_rcpf(1.0f + e);
}

// ---------------------------------------------------------------------------
// Kernel 0: convert + transpose weights to bf16 (N-major so B-fragment loads
// are contiguous 16B runs).
// ---------------------------------------------------------------------------
__global__ void k_convert(const float* __restrict__ Wh,
                          const float* __restrict__ Wqk,
                          const float* __restrict__ Wo,
                          __bf16* __restrict__ WhT,
                          __bf16* __restrict__ WqkT,
                          __bf16* __restrict__ WoT) {
  int idx = blockIdx.x * 256 + threadIdx.x;
  const int NH = 2048 * 512, NQ = 128 * 512, NO = 512 * 1024;
  if (idx < NH) {                       // WhT[n][k], Wh is (512 x 2048)
    int n = idx / 512, k = idx % 512;
    WhT[idx] = (__bf16)Wh[k * 2048 + n];
  } else if (idx < NH + NQ) {           // WqkT[n][k], Wqk is (512 x 128)
    int t = idx - NH; int n = t / 512, k = t % 512;
    WqkT[t] = (__bf16)Wqk[k * 128 + n];
  } else if (idx < NH + NQ + NO) {      // WoT[n][k], Wo is (1024 x 512)
    int t = idx - NH - NQ; int n = t / 1024, k = t % 1024;
    WoT[t] = (__bf16)Wo[k * 512 + n];
  }
}

// ---------------------------------------------------------------------------
// Kernel A: LayerNorm + hid/qk projections with SiLU epilogue.
// One block = one 16-row M-tile.  A-fragments for the whole K=512 row block
// are kept resident in registers (16 frags = 128 VGPRs); 8 waves sweep the
// 136 N-tiles with B-fragment double buffering.
// ---------------------------------------------------------------------------
__global__ __launch_bounds__(256) void k_ln_proj(
    const float* __restrict__ x,  const float* __restrict__ ln_g,
    const float* __restrict__ ln_b,
    const __bf16* __restrict__ WhT, const float* __restrict__ bh,
    const __bf16* __restrict__ WqkT, const float* __restrict__ bqk,
    const float* __restrict__ gamma, const float* __restrict__ beta,
    __bf16* __restrict__ vT, __bf16* __restrict__ gate,
    __bf16* __restrict__ qm, __bf16* __restrict__ km) {
  const int LNST = 520;                       // 512 + pad (16B-aligned rows)
  __shared__ __bf16 ln[16 * LNST];
  __shared__ float red_s[256], red_q[256], s_mu[16], s_rs[16];

  const int t = threadIdx.x;
  const int lane = t & 31, hl = lane >> 4, l15 = lane & 15, w = t >> 5;
  const int b = blockIdx.x >> 7;
  const int s0 = (blockIdx.x & 127) << 4;

  // ---- LayerNorm: 16 threads per row, 32 elements per thread -------------
  {
    const int r = t >> 4, c = t & 15;
    const float* xrow = x + ((size_t)(b * 2048 + s0 + r)) * 512 + c * 32;
    float vals[32];
    float sum = 0.f, sq = 0.f;
#pragma unroll
    for (int i = 0; i < 8; i++) {
      float4 f = ((const float4*)xrow)[i];
      vals[4 * i + 0] = f.x; vals[4 * i + 1] = f.y;
      vals[4 * i + 2] = f.z; vals[4 * i + 3] = f.w;
      sum += f.x + f.y + f.z + f.w;
      sq  += f.x * f.x + f.y * f.y + f.z * f.z + f.w * f.w;
    }
    red_s[t] = sum; red_q[t] = sq;
    __syncthreads();
    if (c == 0) {
      float S1 = 0.f, S2 = 0.f;
      for (int i = 0; i < 16; i++) { S1 += red_s[r * 16 + i]; S2 += red_q[r * 16 + i]; }
      float mu = S1 * (1.f / 512.f);
      float var = S2 * (1.f / 512.f) - mu * mu;
      s_mu[r] = mu; s_rs[r] = rsqrtf(var + 1e-5f);
    }
    __syncthreads();
    float mu = s_mu[r], rs = s_rs[r];
#pragma unroll
    for (int i = 0; i < 32; i++) {
      int k = c * 32 + i;
      ln[r * LNST + k] = (__bf16)((vals[i] - mu) * rs * ln_g[k] + ln_b[k]);
    }
  }
  __syncthreads();

  // ---- A-fragments resident for all N-tiles ------------------------------
  FragB afr[16];
  {
    const __bf16* lrow = ln + l15 * LNST;
#pragma unroll
    for (int kt = 0; kt < 16; kt++) load_a(afr[kt], lrow, kt, hl);
  }

  // ---- hid projection: 128 N-tiles of Wh, uniform v/gate split -----------
  for (int nt = w; nt < 128; nt += 8) {
    const int n = nt * 16 + l15;
    v8f acc;
    {
      const float bias = bh[n];
#pragma unroll
      for (int i = 0; i < 8; i++) acc[i] = bias;
    }
    const __bf16* wrow = WhT + (size_t)n * 512;
    FragB b0; load_b(b0, wrow, 0, hl);
#pragma unroll
    for (int kt = 0; kt < 16; kt++) {
      FragB b1;
      if (kt < 15) load_b(b1, wrow, kt + 1, hl); else b1 = b0;
      acc = wmma_bf16(afr[kt].v, b0.v, acc);
      b0 = b1;
    }
    if (nt < 64) {                 // v half: store transposed (j-contiguous)
      __bf16* vcol = vT + ((size_t)(b * 1024 + n)) * 2048 + s0;
#pragma unroll
      for (int rr = 0; rr < 8; rr++)
        vcol[rr + hl * 8] = (__bf16)fast_silu(acc[rr]);
    } else {                       // gate half: row-major
#pragma unroll
      for (int rr = 0; rr < 8; rr++) {
        const int srow = s0 + rr + hl * 8;
        gate[((size_t)(b * 2048 + srow)) * 1024 + (n - 1024)] =
            (__bf16)fast_silu(acc[rr]);
      }
    }
  }

  // ---- qk projection: one 16-col tile per wave ---------------------------
  {
    const int n = w * 16 + l15;    // 0..127
    v8f acc;
    {
      const float bias = bqk[n];
#pragma unroll
      for (int i = 0; i < 8; i++) acc[i] = bias;
    }
    const __bf16* wrow = WqkT + (size_t)n * 512;
    FragB b0; load_b(b0, wrow, 0, hl);
#pragma unroll
    for (int kt = 0; kt < 16; kt++) {
      FragB b1;
      if (kt < 15) load_b(b1, wrow, kt + 1, hl); else b1 = b0;
      acc = wmma_bf16(afr[kt].v, b0.v, acc);
      b0 = b1;
    }
    const float g0 = gamma[n], g1 = gamma[128 + n];
    const float be0 = beta[n], be1 = beta[128 + n];
#pragma unroll
    for (int rr = 0; rr < 8; rr++) {
      const int srow = s0 + rr + hl * 8;
      const float z = fast_silu(acc[rr]);
      qm[((size_t)(b * 2048 + srow)) * 128 + n] = (__bf16)(z * g0 + be0);
      km[((size_t)(b * 2048 + srow)) * 128 + n] = (__bf16)(z * g1 + be1);
    }
  }
}

// ---------------------------------------------------------------------------
// Kernel B: fused quadratic attention.  Block = 16-row i-tile, 8 waves.
// Per 256-j chunk: each wave builds its own 16x32 P tile (relu(sim/S)^2, bf16
// in LDS), barrier, then all waves run P @ v over their 128 v-columns.
// ---------------------------------------------------------------------------
__global__ __launch_bounds__(256) void k_attn(
    const __bf16* __restrict__ qm, const __bf16* __restrict__ km,
    const __bf16* __restrict__ vT, const __bf16* __restrict__ gate,
    __bf16* __restrict__ Vg) {
  const int PST = 264;                       // 256 + pad (16B-aligned rows)
  __shared__ __bf16 P[16 * PST];

  const int t = threadIdx.x;
  const int lane = t & 31, hl = lane >> 4, l15 = lane & 15, w = t >> 5;
  const int b = blockIdx.x >> 7;
  const int i0 = (blockIdx.x & 127) << 4;
  const float invS2 = 1.0f / (2048.0f * 2048.0f);  // relu(sim)^2 / S^2

  // Resident q A-fragments for this i-tile (K = 128 -> 4 frags)
  FragB qa[4];
  {
    const __bf16* qrow = qm + ((size_t)(b * 2048 + i0 + l15)) * 128;
#pragma unroll
    for (int kt = 0; kt < 4; kt++) load_a(qa[kt], qrow, kt, hl);
  }

  v8f acc[8];                                // 16 rows x 128 cols f32
#pragma unroll
  for (int i = 0; i < 8; i++) {
#pragma unroll
    for (int j = 0; j < 8; j++) acc[i][j] = 0.0f;
  }

  const __bf16* vbase = vT + ((size_t)(b * 1024 + w * 128 + l15)) * 2048;

  for (int jc = 0; jc < 8; jc++) {           // 2048 j in chunks of 256
    const int jbase = jc * 256;
    if (jc + 1 < 8) {                        // stream next chunk's k rows
      __builtin_prefetch(
          km + ((size_t)(b * 2048 + jbase + 256 + w * 32 + l15)) * 128, 0, 1);
    }
    // ---- phase 1: this wave's 16x32 P tile -------------------------------
#pragma unroll
    for (int jt2 = 0; jt2 < 2; jt2++) {
      const int j0 = jbase + w * 32 + jt2 * 16;
      const __bf16* krow = km + ((size_t)(b * 2048 + j0 + l15)) * 128;
      FragB kb[4];
#pragma unroll
      for (int kt = 0; kt < 4; kt++) load_b(kb[kt], krow, kt, hl);
      v8f sacc;
#pragma unroll
      for (int i = 0; i < 8; i++) sacc[i] = 0.0f;
#pragma unroll
      for (int kt = 0; kt < 4; kt++)
        sacc = wmma_bf16(qa[kt].v, kb[kt].v, sacc);
#pragma unroll
      for (int rr = 0; rr < 8; rr++) {
        float sv = fmaxf(sacc[rr], 0.0f);
        sv = sv * sv * invS2;
        P[(rr + hl * 8) * PST + w * 32 + jt2 * 16 + l15] = (__bf16)sv;
      }
    }
    __syncthreads();
    // ---- phase 2: P(16x256) @ v(256 x this wave's 128 cols) --------------
#pragma unroll 2
    for (int jb = 0; jb < 8; jb++) {
      FragB pa;
      {
        const __bf16* prow = P + l15 * PST + jb * 32;
        pa.h[0] = *(const bf16x8*)(prow + hl * 8);
        pa.h[1] = *(const bf16x8*)(prow + 16 + hl * 8);
      }
      const __bf16* vrow0 = vbase + jbase + jb * 32;
      FragB vb; load_b(vb, vrow0, 0, hl);          // kt arg 0: offset 0
#pragma unroll
      for (int nt = 0; nt < 8; nt++) {
        FragB vn;
        if (nt < 7) load_b(vn, vrow0 + (size_t)(nt + 1) * 16 * 2048, 0, hl);
        else vn = vb;
        acc[nt] = wmma_bf16(pa.v, vb.v, acc[nt]);
        vb = vn;
      }
    }
    __syncthreads();                         // before P is overwritten
  }

  // ---- epilogue: gate multiply, store bf16 V -----------------------------
#pragma unroll
  for (int nt = 0; nt < 8; nt++) {
    const int n = w * 128 + nt * 16 + l15;
#pragma unroll
    for (int rr = 0; rr < 8; rr++) {
      const int i = i0 + rr + hl * 8;
      const size_t idx = ((size_t)(b * 2048 + i)) * 1024 + n;
      const float g = (float)gate[idx];
      Vg[idx] = (__bf16)(acc[nt][rr] * g);
    }
  }
}

// ---------------------------------------------------------------------------
// Kernel C: out = Vg(16384x1024) @ Wo + bo + x   (f32 out, residual add)
// ---------------------------------------------------------------------------
__global__ __launch_bounds__(256) void k_out(
    const __bf16* __restrict__ Vg, const __bf16* __restrict__ WoT,
    const float* __restrict__ bo, const float* __restrict__ x,
    float* __restrict__ out) {
  const int t = threadIdx.x;
  const int lane = t & 31, hl = lane >> 4, l15 = lane & 15, w = t >> 5;
  const int b = blockIdx.x >> 7;
  const int i0 = (blockIdx.x & 127) << 4;
  const int n0 = w * 64;

  v8f acc[4];
#pragma unroll
  for (int nt = 0; nt < 4; nt++) {
    const float bias = bo[n0 + nt * 16 + l15];
#pragma unroll
    for (int i = 0; i < 8; i++) acc[nt][i] = bias;
  }

  const __bf16* arow = Vg + ((size_t)(b * 2048 + i0 + l15)) * 1024;
  FragB a; load_a(a, arow, 0, hl);
#pragma unroll 4
  for (int kt = 0; kt < 32; kt++) {
    FragB an;
    if (kt < 31) load_a(an, arow, kt + 1, hl); else an = a;
    FragB bb[4];
#pragma unroll
    for (int nt = 0; nt < 4; nt++) {
      const __bf16* wrow = WoT + (size_t)(n0 + nt * 16 + l15) * 1024;
      load_b(bb[nt], wrow, kt, hl);
    }
#pragma unroll
    for (int nt = 0; nt < 4; nt++)
      acc[nt] = wmma_bf16(a.v, bb[nt].v, acc[nt]);
    a = an;
  }

#pragma unroll
  for (int nt = 0; nt < 4; nt++) {
    const int n = n0 + nt * 16 + l15;
#pragma unroll
    for (int rr = 0; rr < 8; rr++) {
      const int i = i0 + rr + hl * 8;
      const size_t idx = ((size_t)(b * 2048 + i)) * 512 + n;
      out[idx] = acc[nt][rr] + x[idx];
    }
  }
}

// ---------------------------------------------------------------------------
extern "C" void kernel_launch(void* const* d_in, const int* in_sizes, int n_in,
                              void* d_out, int out_size, void* d_ws, size_t ws_size,
                              hipStream_t stream) {
  (void)in_sizes; (void)n_in; (void)out_size; (void)ws_size;
  const float* x     = (const float*)d_in[0];
  const float* ln_g  = (const float*)d_in[1];
  const float* ln_b  = (const float*)d_in[2];
  const float* Wh    = (const float*)d_in[3];
  const float* bh    = (const float*)d_in[4];
  const float* Wqk   = (const float*)d_in[5];
  const float* bqk   = (const float*)d_in[6];
  const float* gamma = (const float*)d_in[7];
  const float* beta  = (const float*)d_in[8];
  const float* Wo    = (const float*)d_in[9];
  const float* bo    = (const float*)d_in[10];
  float* out = (float*)d_out;

  char* ws = (char*)d_ws;
  size_t off = 0;
  auto wsalloc = [&](size_t bytes) -> void* {
    void* p = ws + off;
    off += (bytes + 255) & ~(size_t)255;
    return p;
  };
  __bf16* WhT  = (__bf16*)wsalloc(2048ull * 512 * 2);
  __bf16* WqkT = (__bf16*)wsalloc(128ull * 512 * 2);
  __bf16* WoT  = (__bf16*)wsalloc(512ull * 1024 * 2);
  __bf16* qb   = (__bf16*)wsalloc(8ull * 2048 * 128 * 2);
  __bf16* kb   = (__bf16*)wsalloc(8ull * 2048 * 128 * 2);
  __bf16* vT   = (__bf16*)wsalloc(8ull * 1024 * 2048 * 2);
  __bf16* gate = (__bf16*)wsalloc(8ull * 2048 * 1024 * 2);
  __bf16* Vg   = (__bf16*)wsalloc(8ull * 2048 * 1024 * 2);

  const int convN = 2048 * 512 + 128 * 512 + 512 * 1024;
  k_convert<<<(convN + 255) / 256, 256, 0, stream>>>(Wh, Wqk, Wo, WhT, WqkT, WoT);
  k_ln_proj<<<1024, 256, 0, stream>>>(x, ln_g, ln_b, WhT, bh, WqkT, bqk,
                                      gamma, beta, vT, gate, qb, kb);
  k_attn<<<1024, 256, 0, stream>>>(qb, kb, vT, gate, Vg);
  k_out<<<1024, 256, 0, stream>>>(Vg, WoT, bo, x, out);
}
